// LSeqSleepNet_8873402433773
// MI455X (gfx1250) — matrix operand order repaired and checked by reading, not compiled, not run on hardware
//
#include <hip/hip_runtime.h>
#include <hip/hip_bf16.h>

#define BB    1024
#define TT    100
#define DD    256
#define HH    128
#define AATT  64
#define FOURH 512
#define EPSF  1e-5f

#if defined(__has_builtin)
#  if __has_builtin(__builtin_amdgcn_global_load_async_to_lds_b128) && \
      __has_builtin(__builtin_amdgcn_s_wait_asynccnt)
#    define USE_ASYNC_LDS 1
#  endif
#endif
#ifndef USE_ASYNC_LDS
#  define USE_ASYNC_LDS 0
#endif

typedef __attribute__((ext_vector_type(16))) _Float16 v16h;
typedef __attribute__((ext_vector_type(8)))  float    v8f;
typedef __attribute__((ext_vector_type(4)))  int      v4i;

#if USE_ASYNC_LDS
typedef __attribute__((address_space(1))) v4i* gptr_v4i;
typedef __attribute__((address_space(3))) v4i* lptr_v4i;
// per-lane 16B async copy global -> LDS (GLOBAL_LOAD_ASYNC_TO_LDS_B128, ASYNCcnt)
__device__ __forceinline__ void async_cp16(const _Float16* g, _Float16* l) {
    __builtin_amdgcn_global_load_async_to_lds_b128((gptr_v4i)(g), (lptr_v4i)(l), 0, 0);
}
#endif

__device__ __forceinline__ float sigmf(float x) { return 1.0f / (1.0f + __expf(-x)); }

__global__ void zero_kernel(float* p, int n) {
    int i = blockIdx.x * blockDim.x + threadIdx.x;
    if (i < n) p[i] = 0.0f;
}

// fp32 -> f16 elementwise
__global__ void cvt_f16(const float* __restrict__ src, _Float16* __restrict__ dst, int n) {
    int i = blockIdx.x * blockDim.x + threadIdx.x;
    if (i < n) dst[i] = (_Float16)src[i];
}

// W[K,N] fp32 -> Wt[N,K] f16 (one-time, makes B-fragment gathers contiguous)
__global__ void cvt_w_t(const float* __restrict__ W, _Float16* __restrict__ Wt, int K, int N) {
    int i = blockIdx.x * blockDim.x + threadIdx.x;
    if (i >= N * K) return;
    int n = i / K, k = i - n * K;
    Wt[i] = (_Float16)W[(size_t)k * N + n];
}

// C[M,N] = A[M,K] @ B[K,N]; A f16 row-major (lda), B pre-transposed f16 [N,K].
// Block: 256 threads = 8 waves; block tile 64(M) x 128(N).
// Wave (wm 0..3, wn 0..1): 16x64 output = 4 WMMA tiles, A-fragment reused x4.
// Double-buffered LDS; next chunk staged via async global->LDS copies (or
// register staging pinned with sched_barrier) while current chunk's WMMAs run.
template <int K>
__global__ __launch_bounds__(256)
void wmma_gemm(const _Float16* __restrict__ A, int lda,
               const _Float16* __restrict__ Bt,
               float* __restrict__ C, int ldc)
{
    constexpr int NCHUNK = K / 32;
    __shared__ _Float16 As[2][64][40];    // [buf][m][k], 80B row stride (16B aligned)
    __shared__ _Float16 Bs[2][128][40];   // [buf][n][k]

    const int tid  = threadIdx.x;
    const int m0   = blockIdx.y * 64;
    const int n0   = blockIdx.x * 128;
    const int wave = tid >> 5;
    const int lane = tid & 31;
    const int wm   = wave >> 1;          // 0..3
    const int wn   = wave & 1;           // 0..1
    const int l15  = lane & 15;
    const bool hi  = lane >= 16;

    const int ar = tid >> 2, acg = (tid & 3) * 8;   // staging: 64 rows x 4x(8 halves)

    // hoisted global staging pointers (this thread's slice)
    const _Float16* aSrc  = &A[(size_t)(m0 + ar) * lda + acg];
    const _Float16* bSrc0 = &Bt[(size_t)(n0 + ar) * K + acg];
    const _Float16* bSrc1 = &Bt[(size_t)(n0 + 64 + ar) * K + acg];

#if USE_ASYNC_LDS
    // preload chunk 0 via async DMA
    async_cp16(aSrc,  &As[0][ar][acg]);
    async_cp16(bSrc0, &Bs[0][ar][acg]);
    async_cp16(bSrc1, &Bs[0][64 + ar][acg]);
    __builtin_amdgcn_s_wait_asynccnt(0);
    __syncthreads();
#else
    uint4 ta  = *reinterpret_cast<const uint4*>(aSrc);
    uint4 tb0 = *reinterpret_cast<const uint4*>(bSrc0);
    uint4 tb1 = *reinterpret_cast<const uint4*>(bSrc1);
    *reinterpret_cast<uint4*>(&As[0][ar][acg])      = ta;
    *reinterpret_cast<uint4*>(&Bs[0][ar][acg])      = tb0;
    *reinterpret_cast<uint4*>(&Bs[0][64 + ar][acg]) = tb1;
    __syncthreads();
#endif

    v8f acc[4] = {};

#pragma unroll
    for (int c = 0; c < NCHUNK; ++c) {
        const int buf = c & 1;
        // issue NEXT chunk's staging early (latency hidden under WMMAs below)
        if (c + 1 < NCHUNK) {
            const int kn = (c + 1) * 32;
            const int nb = buf ^ 1;
#if USE_ASYNC_LDS
            // direct global->LDS DMA into the other buffer (safe: buf^1 reads
            // finished at the barrier ending chunk c-1)
            async_cp16(aSrc + kn,  &As[nb][ar][acg]);
            async_cp16(bSrc0 + kn, &Bs[nb][ar][acg]);
            async_cp16(bSrc1 + kn, &Bs[nb][64 + ar][acg]);
#else
            ta  = *reinterpret_cast<const uint4*>(aSrc + kn);
            tb0 = *reinterpret_cast<const uint4*>(bSrc0 + kn);
            tb1 = *reinterpret_cast<const uint4*>(bSrc1 + kn);
            __builtin_amdgcn_sched_barrier(0);   // pin loads above the WMMAs
#endif
            if (c + 2 < NCHUNK) {   // global_prefetch_b8 two chunks ahead
                __builtin_prefetch(aSrc + (c + 2) * 32);
                __builtin_prefetch(bSrc0 + (c + 2) * 32);
                __builtin_prefetch(bSrc1 + (c + 2) * 32);
            }
        }

        // A fragment (contiguous K order, matching B):
        // lanes 0-15: M=lane, elems 0..15 -> K 0..15
        // lanes 16-31: M=lane-16, elems 0..15 -> K 16..31
        v16h af;
        const _Float16* ap = &As[buf][wm * 16 + l15][0];
        const int ko = hi ? 16 : 0;
#pragma unroll
        for (int j = 0; j < 16; ++j) af[j] = ap[ko + j];
        // 4 B fragments (lane = column; lanes 0-15 K 0..15, lanes 16-31 K 16..31)
#pragma unroll
        for (int nt = 0; nt < 4; ++nt) {
            v16h bf;
            const _Float16* bp = &Bs[buf][wn * 64 + nt * 16 + l15][0];
#pragma unroll
            for (int j = 0; j < 16; ++j) bf[j] = bp[ko + j];
            acc[nt] = __builtin_amdgcn_wmma_f32_16x16x32_f16(false, af, false, bf,
                                                             (short)0, acc[nt], false, false);
        }

        if (c + 1 < NCHUNK) {
#if USE_ASYNC_LDS
            __builtin_amdgcn_s_wait_asynccnt(0);   // own copies done; barrier covers peers
            __syncthreads();
#else
            const int nb = buf ^ 1;
            *reinterpret_cast<uint4*>(&As[nb][ar][acg])      = ta;
            *reinterpret_cast<uint4*>(&Bs[nb][ar][acg])      = tb0;
            *reinterpret_cast<uint4*>(&Bs[nb][64 + ar][acg]) = tb1;
            __syncthreads();
#endif
        }
    }

    // C 16x16 layout: VGPR r -> M = r (lanes 0-15) / 8+r (lanes 16-31); N = lane&15
    const int crow = m0 + wm * 16 + (hi ? 8 : 0);
#pragma unroll
    for (int nt = 0; nt < 4; ++nt) {
        const int ccol = n0 + wn * 64 + nt * 16 + l15;
#pragma unroll
        for (int r = 0; r < 8; ++r) C[(size_t)(crow + r) * ldc + ccol] = acc[nt][r];
    }
}

// Per-column mean/var over M rows for BOTH gate matrices in one launch.
// grid = 16 blocks: 0..7 -> gx, 8..15 -> gh. 256 threads = 64 cols x 4 row-groups.
__global__ __launch_bounds__(256)
void gate_stats(const float* __restrict__ gx, float2* __restrict__ ox,
                const float* __restrict__ gh, float2* __restrict__ oh, int M)
{
    const float* X = (blockIdx.x < 8) ? gx : gh;
    float2* out    = (blockIdx.x < 8) ? ox : oh;
    int blk = blockIdx.x & 7;
    int c = threadIdx.x & 63, g = threadIdx.x >> 6;
    int j = blk * 64 + c;
    float s = 0.0f, ss = 0.0f;
    for (int r = g; r < M; r += 4) {
        float v = X[(size_t)r * FOURH + j];
        s += v; ss += v * v;
    }
    __shared__ float sh[2][4][64];
    sh[0][g][c] = s; sh[1][g][c] = ss;
    __syncthreads();
    if (g == 0) {
        s  = sh[0][0][c] + sh[0][1][c] + sh[0][2][c] + sh[0][3][c];
        ss = sh[1][0][c] + sh[1][1][c] + sh[1][2][c] + sh[1][3][c];
        float mean = s * (1.0f / M);
        out[j] = make_float2(mean, ss * (1.0f / M) - mean * mean);
    }
}

// Per-column mean/var, generic (used for cell state, N=128 -> grid 2)
__global__ __launch_bounds__(256)
void col_stats(const float* __restrict__ X, float2* __restrict__ out, int M, int ldx)
{
    int c = threadIdx.x & 63, g = threadIdx.x >> 6;
    int j = blockIdx.x * 64 + c;
    float s = 0.0f, ss = 0.0f;
    for (int r = g; r < M; r += 4) {
        float v = X[(size_t)r * ldx + j];
        s += v; ss += v * v;
    }
    __shared__ float sh[2][4][64];
    sh[0][g][c] = s; sh[1][g][c] = ss;
    __syncthreads();
    if (g == 0) {
        s  = sh[0][0][c] + sh[0][1][c] + sh[0][2][c] + sh[0][3][c];
        ss = sh[1][0][c] + sh[1][1][c] + sh[1][2][c] + sh[1][3][c];
        float mean = s * (1.0f / M);
        out[j] = make_float2(mean, ss * (1.0f / M) - mean * mean);
    }
}

__global__ __launch_bounds__(256)
void cell_update(const float* __restrict__ gx, const float* __restrict__ gh,
                 const float2* __restrict__ stx, const float2* __restrict__ sth,
                 const float* __restrict__ s_xh, const float* __restrict__ s_hh,
                 const float* __restrict__ o_hh,
                 float* __restrict__ c, float* __restrict__ sig_o)
{
    int idx = blockIdx.x * blockDim.x + threadIdx.x;   // < B*H
    int b = idx >> 7;
    int h = idx & 127;
    float gate[4];
#pragma unroll
    for (int q = 0; q < 4; ++q) {
        int col = q * HH + h;
        float zx = gx[(size_t)b * FOURH + col];
        float zh = gh[(size_t)b * FOURH + col];
        float2 sx = stx[col], sv = sth[col];
        float nx = s_xh[col] * (zx - sx.x) * rsqrtf(sx.y + EPSF);
        float nh = s_hh[col] * (zh - sv.x) * rsqrtf(sv.y + EPSF) + o_hh[col];
        gate[q] = nx + nh;
    }
    float f  = sigmf(gate[0]);
    float ig = sigmf(gate[1]);
    float og = sigmf(gate[2]);
    float jg = tanhf(gate[3]);
    float nc = c[idx] * f + ig * jg;
    c[idx]     = nc;
    sig_o[idx] = og;
}

__global__ __launch_bounds__(256)
void h_step(const float* __restrict__ c, const float2* __restrict__ stc,
            const float* __restrict__ s_c, const float* __restrict__ o_c,
            const float* __restrict__ sig_o,
            _Float16* __restrict__ h16, float* __restrict__ hs_t)
{
    int idx = blockIdx.x * blockDim.x + threadIdx.x;
    int h = idx & 127;
    float2 st = stc[h];
    float v = tanhf(s_c[h] * (c[idx] - st.x) * rsqrtf(st.y + EPSF) + o_c[h]) * sig_o[idx];
    h16[idx] = (_Float16)v;   // f16 operand for next step's WMMA GEMM
    hs_t[idx] = v;            // fp32 history for attention
}

// scores[b,t] = sum_a tanh(hcat[b,t,:] @ Wo[:,a] + bo[a]) * uo[a]
__global__ __launch_bounds__(64)
void attn_scores(const float* __restrict__ hsf, const float* __restrict__ hsb,
                 const float* __restrict__ Wo, const float* __restrict__ bo,
                 const float* __restrict__ uo, float* __restrict__ scores)
{
    int bt = blockIdx.x;
    int t = bt % TT, b = bt / TT;
    int tid = threadIdx.x;
    __shared__ float hsm[2 * HH];
    for (int i = tid; i < 2 * HH; i += 64)
        hsm[i] = (i < HH) ? hsf[((size_t)t * BB + b) * HH + i]
                          : hsb[((size_t)t * BB + b) * HH + (i - HH)];
    __syncthreads();
    float acc = bo[tid];
    for (int hh = 0; hh < 2 * HH; ++hh) acc += hsm[hh] * Wo[hh * AATT + tid];
    float contrib = tanhf(acc) * uo[tid];
    __shared__ float red[64];
    red[tid] = contrib;
    __syncthreads();
    for (int s = 32; s > 0; s >>= 1) {
        if (tid < s) red[tid] += red[tid + s];
        __syncthreads();
    }
    if (tid == 0) scores[bt] = red[0];
}

// softmax over T + weighted sum; grid = B blocks, 256 threads (one per output dim)
__global__ __launch_bounds__(256)
void attn_pool(const float* __restrict__ scores, const float* __restrict__ hsf,
               const float* __restrict__ hsb, float* __restrict__ out)
{
    int b = blockIdx.x, d = threadIdx.x;
    __shared__ float sc[TT];
    for (int i = d; i < TT; i += 256) sc[i] = scores[(size_t)b * TT + i];
    __syncthreads();
    float m = -1e30f;
    for (int t = 0; t < TT; ++t) m = fmaxf(m, sc[t]);
    float s = 0.0f;
    for (int t = 0; t < TT; ++t) s += __expf(sc[t] - m);
    float inv = 1.0f / s;
    float acc = 0.0f;
    for (int t = 0; t < TT; ++t) {
        float al = __expf(sc[t] - m) * inv;
        float hv = (d < HH) ? hsf[((size_t)t * BB + b) * HH + d]
                            : hsb[((size_t)t * BB + b) * HH + (d - HH)];
        acc += al * hv;
    }
    out[(size_t)b * (2 * HH) + d] = acc;
}

extern "C" void kernel_launch(void* const* d_in, const int* in_sizes, int n_in,
                              void* d_out, int out_size, void* d_ws, size_t ws_size,
                              hipStream_t stream)
{
    (void)in_sizes; (void)n_in; (void)out_size; (void)ws_size;
    const float* x = (const float*)d_in[0];
    const float* W_xh[2] = {(const float*)d_in[1], (const float*)d_in[8]};
    const float* W_hh[2] = {(const float*)d_in[2], (const float*)d_in[9]};
    const float* s_xh[2] = {(const float*)d_in[3], (const float*)d_in[10]};
    const float* s_hh[2] = {(const float*)d_in[4], (const float*)d_in[11]};
    const float* o_hh[2] = {(const float*)d_in[5], (const float*)d_in[12]};
    const float* s_c[2]  = {(const float*)d_in[6], (const float*)d_in[13]};
    const float* o_c[2]  = {(const float*)d_in[7], (const float*)d_in[14]};
    const float* Wo = (const float*)d_in[15];
    const float* bo = (const float*)d_in[16];
    const float* uo = (const float*)d_in[17];
    float* out = (float*)d_out;

    float* ws = (float*)d_ws;
    size_t off = 0;
    auto alloc = [&](size_t nfloats) -> float* {
        float* p = ws + off;
        off += (nfloats + 63) & ~(size_t)63;   // 256-byte aligned chunks
        return p;
    };
    float*  gx    = alloc((size_t)BB * FOURH);
    float*  gh    = alloc((size_t)BB * FOURH);
    float2* stx   = (float2*)alloc((size_t)FOURH * 2);
    float2* sth   = (float2*)alloc((size_t)FOURH * 2);
    float2* stc   = (float2*)alloc((size_t)HH * 2);
    float*  cbuf  = alloc((size_t)BB * HH);
    float*  sig_o = alloc((size_t)BB * HH);
    float*  hs[2];
    hs[0] = alloc((size_t)TT * BB * HH);
    hs[1] = alloc((size_t)TT * BB * HH);
    float* scores = alloc((size_t)BB * TT);
    // f16 operand pools (allocated in float units, 2 halves per float)
    _Float16* xh      = (_Float16*)alloc((size_t)BB * TT * DD / 2);
    _Float16* h16     = (_Float16*)alloc((size_t)BB * HH / 2);
    _Float16* Wxh_t[2] = {(_Float16*)alloc((size_t)DD * FOURH / 2),
                          (_Float16*)alloc((size_t)DD * FOURH / 2)};
    _Float16* Whh_t[2] = {(_Float16*)alloc((size_t)HH * FOURH / 2),
                          (_Float16*)alloc((size_t)HH * FOURH / 2)};

    // One-time conversions: x -> f16, weights -> f16 transposed [N][K]
    {
        int nx = BB * TT * DD;
        cvt_f16<<<(nx + 255) / 256, 256, 0, stream>>>(x, xh, nx);
        for (int dir = 0; dir < 2; ++dir) {
            cvt_w_t<<<(FOURH * DD + 255) / 256, 256, 0, stream>>>(W_xh[dir], Wxh_t[dir], DD, FOURH);
            cvt_w_t<<<(FOURH * HH + 255) / 256, 256, 0, stream>>>(W_hh[dir], Whh_t[dir], HH, FOURH);
        }
    }

    dim3 gGemm(FOURH / 128, BB / 64);   // (4, 16)

    for (int dir = 0; dir < 2; ++dir) {
        zero_kernel<<<(BB * HH + 255) / 256, 256, 0, stream>>>(cbuf, BB * HH);
        zero_kernel<<<(BB * HH / 2 + 255) / 256, 256, 0, stream>>>((float*)h16, BB * HH / 2);
        for (int s = 0; s < TT; ++s) {
            int t = (dir == 0) ? s : (TT - 1 - s);
            // gates_x = x[:,t,:] @ W_xh   (A rows stride T*D halves)
            wmma_gemm<DD><<<gGemm, 256, 0, stream>>>(xh + (size_t)t * DD, TT * DD,
                                                     Wxh_t[dir], gx, FOURH);
            // gates_h = h @ W_hh
            wmma_gemm<HH><<<gGemm, 256, 0, stream>>>(h16, HH, Whh_t[dir], gh, FOURH);
            gate_stats<<<16, 256, 0, stream>>>(gx, stx, gh, sth, BB);
            cell_update<<<(BB * HH) / 256, 256, 0, stream>>>(gx, gh, stx, sth,
                                                             s_xh[dir], s_hh[dir], o_hh[dir],
                                                             cbuf, sig_o);
            col_stats<<<HH / 64, 256, 0, stream>>>(cbuf, stc, BB, HH);
            h_step<<<(BB * HH) / 256, 256, 0, stream>>>(cbuf, stc, s_c[dir], o_c[dir],
                                                        sig_o, h16,
                                                        hs[dir] + (size_t)t * BB * HH);
        }
    }
    attn_scores<<<BB * TT, 64, 0, stream>>>(hs[0], hs[1], Wo, bo, uo, scores);
    attn_pool<<<BB, 256, 0, stream>>>(scores, hs[0], hs[1], out);
}
